// EmulatedDMoE_23433341567172
// MI455X (gfx1250) — compile-verified, matrix operand order checked by reading, not run on hardware
//
#include <hip/hip_runtime.h>
#include <hip/hip_bf16.h>
#include <cstdint>

// ---------------------------------------------------------------------------
// MoE (top-2 of 16) routed implementation for gfx1250.
//  B=2048 D=1024 E=16 H=4096.  Bandwidth-bound (~620MB streamed ~= 27us at
//  23.3TB/s); compute (68.7 GFLOP) via v_wmma_f32_16x16x32_bf16.
//  Double-buffered LDS, one barrier per K-step so cvt/ds-stores co-execute
//  with XDL WMMA.  Stage-B A-tiles via async global->LDS DMA (ASYNCcnt).
// ---------------------------------------------------------------------------

#define B_ 2048
#define D_ 1024
#define E_ 16
#define H_ 4096

#define TM 32      // rows per block tile (2 waves of 16 in M)
#define TN 128     // cols per block tile (4 waves x 32 in N)
#define TK 64      // K per LDS stage = 2 WMMA sub-steps
#define LDT 72     // padded LDS stride (elems); 72*2B = 144B (16B aligned rows)

typedef __attribute__((ext_vector_type(16))) __bf16 v16bf;
typedef __attribute__((ext_vector_type(8)))  __bf16 v8bf;
typedef __attribute__((ext_vector_type(4)))  __bf16 v4bf;
typedef __attribute__((ext_vector_type(8)))  float  v8f;

static __device__ __forceinline__ int imin(int a, int b) { return a < b ? a : b; }

// Async DMA: 16 bytes global -> LDS per lane (CDNA5 GLOBAL_LOAD_ASYNC_TO_LDS).
static __device__ __forceinline__ void async_copy_b128(uint32_t lds_off,
                                                       uint64_t gaddr) {
  asm volatile("global_load_async_to_lds_b128 %0, %1, off"
               :: "v"(lds_off), "v"(gaddr) : "memory");
}
static __device__ __forceinline__ void wait_async() {
  asm volatile("s_wait_asynccnt 0x0" ::: "memory");
}

// Load a 16-element bf16 WMMA fragment from an LDS row laid out
// [idx][k0..k63] with stride LDT.  Lane mapping per ISA 7.12.2:
// lanes 0-15 -> K {koff+0..7, 16+koff..23}, koff=8 for lanes 16-31.
static __device__ __forceinline__ v16bf load_frag(const __bf16* rowp, int koff) {
  v8bf lo = *(const v8bf*)(rowp + koff);
  v8bf hi = *(const v8bf*)(rowp + 16 + koff);
  return __builtin_shufflevector(lo, hi, 0, 1, 2, 3, 4, 5, 6, 7,
                                         8, 9, 10, 11, 12, 13, 14, 15);
}

// ---------------------------------------------------------------------------
__global__ void zero_kernel(float* __restrict__ out, int n, int* __restrict__ counts) {
  int i = blockIdx.x * blockDim.x + threadIdx.x;
  for (; i < n; i += gridDim.x * blockDim.x) out[i] = 0.0f;
  if (blockIdx.x == 0 && threadIdx.x < E_) counts[threadIdx.x] = 0;
}

// ---------------------------------------------------------------------------
// Gating: one block (256 threads) per token.
__global__ __launch_bounds__(256)
void gate_kernel(const float* __restrict__ x,
                 const float* __restrict__ gamma,
                 const float* __restrict__ beta,
                 const float* __restrict__ keys,
                 int* __restrict__ counts,
                 int* __restrict__ tokslot,
                 float* __restrict__ gatearr) {
  const int t = blockIdx.x;
  const int tid = threadIdx.x;
  __shared__ float red[256];
  __shared__ float logits[E_];
  __shared__ float s_mu, s_rstd;

  const float* xr = x + (size_t)t * D_;
  float xv[4];
  float s = 0.f, ss = 0.f;
#pragma unroll
  for (int j = 0; j < 4; ++j) {
    float v = xr[tid + 256 * j];
    xv[j] = v;
    s += v;
    ss += v * v;
  }
  red[tid] = s;
  __syncthreads();
  for (int o = 128; o > 0; o >>= 1) {
    if (tid < o) red[tid] += red[tid + o];
    __syncthreads();
  }
  if (tid == 0) s_mu = red[0] * (1.0f / D_);
  __syncthreads();
  red[tid] = ss;
  __syncthreads();
  for (int o = 128; o > 0; o >>= 1) {
    if (tid < o) red[tid] += red[tid + o];
    __syncthreads();
  }
  if (tid == 0) {
    float mu = s_mu;
    float var = red[0] * (1.0f / D_) - mu * mu;
    s_rstd = rsqrtf(var + 1e-5f);
  }
  if (tid < E_) logits[tid] = 0.f;
  __syncthreads();

  const float mu = s_mu, rstd = s_rstd;
  float acc[E_];
#pragma unroll
  for (int e = 0; e < E_; ++e) acc[e] = 0.f;
#pragma unroll
  for (int j = 0; j < 4; ++j) {
    int d = tid + 256 * j;
    float xn = (xv[j] - mu) * rstd * gamma[d] + beta[d];
    const float* kr = keys + (size_t)d * E_;
    float kk[E_];
    float nn = 0.f;
#pragma unroll
    for (int e = 0; e < E_; ++e) { kk[e] = kr[e]; nn += kk[e] * kk[e]; }
    float nrm = fmaxf(sqrtf(nn), 1e-12f);
    float sc = xn / nrm;
#pragma unroll
    for (int e = 0; e < E_; ++e) acc[e] += sc * kk[e];
  }
#pragma unroll
  for (int e = 0; e < E_; ++e) atomicAdd(&logits[e], acc[e]);
  __syncthreads();

  if (tid == 0) {
    int e0 = 0;
    float v0 = logits[0];
    for (int e = 1; e < E_; ++e)
      if (logits[e] > v0) { v0 = logits[e]; e0 = e; }
    int e1 = (e0 == 0) ? 1 : 0;
    float v1 = logits[e1];
    for (int e = 0; e < E_; ++e)
      if (e != e0 && logits[e] > v1) { v1 = logits[e]; e1 = e; }
    float g0 = 1.0f / (1.0f + __expf(v1 - v0));
    float g1 = 1.0f - g0;
    int r0 = atomicAdd(&counts[e0], 1);
    tokslot[e0 * B_ + r0] = t * 2 + 0;
    gatearr[e0 * B_ + r0] = g0;
    int r1 = atomicAdd(&counts[e1], 1);
    tokslot[e1 * B_ + r1] = t * 2 + 1;
    gatearr[e1 * B_ + r1] = g1;
  }
}

// ---------------------------------------------------------------------------
// Stage A: h[(tok*2+slot)][0..H) = relu(Xg @ W1[e] + b1[e])   (bf16 out)
// 256 threads = 8 waves -> 32x128 tile, TK=64.  Double-buffered LDS,
// single barrier per K-step: WMMA on buf[cur] co-executes with cvt+store
// into buf[cur^1].
__global__ __launch_bounds__(256)
void expert_up_kernel(const float* __restrict__ x,
                      const float* __restrict__ W1,
                      const float* __restrict__ b1,
                      const int* __restrict__ counts,
                      const int* __restrict__ tokslot,
                      unsigned short* __restrict__ h_raw) {
  const int e = blockIdx.z;
  const int cnt = counts[e];
  const int rowBase = blockIdx.y * TM;
  if (rowBase >= cnt) return;
  const int nBase = blockIdx.x * TN;
  const int tid = threadIdx.x;
  const int lane = tid & 31, wave = tid >> 5;
  const int wm = wave >> 2, wn = wave & 3;

  __bf16* __restrict__ h = (__bf16*)h_raw;
  __shared__ __attribute__((aligned(16))) __bf16 As[2][TM * LDT];
  __shared__ __attribute__((aligned(16))) __bf16 Bs[2][TN * LDT];
  __shared__ int s_row[TM];

  if (tid < TM) s_row[tid] = tokslot[e * B_ + imin(rowBase + tid, cnt - 1)];
  __syncthreads();

  v8f acc0 = {}, acc1 = {};
  const float* __restrict__ W = W1 + (size_t)e * D_ * H_;

  // A staging: thread -> (row ar, 8 consecutive k at ak)
  const int ar = tid >> 3;
  const int ak = (tid & 7) * 8;
  const float* __restrict__ xrow = x + (size_t)(s_row[ar] >> 1) * D_;
  // B staging: thread -> 4 consecutive k at kb, 8 consecutive n at bn
  const int kb = (tid >> 4) * 4;
  const int bn = (tid & 15) * 8;

  float areg[8];
  float wreg[4][8];

  auto load_stage = [&](int k0) {
#pragma unroll
    for (int j = 0; j < 8; ++j) areg[j] = xrow[k0 + ak + j];
#pragma unroll
    for (int r = 0; r < 4; ++r) {
      const float* src = W + (size_t)(k0 + kb + r) * H_ + nBase + bn;
#pragma unroll
      for (int j = 0; j < 8; ++j) wreg[r][j] = src[j];
    }
    if (k0 + 2 * TK < D_)
      __builtin_prefetch(W + (size_t)(k0 + TK + kb) * H_ + nBase + bn, 0, 3);
  };
  auto store_stage = [&](int buf) {
    v8bf pa;
#pragma unroll
    for (int j = 0; j < 8; ++j) pa[j] = (__bf16)areg[j];
    *(v8bf*)&As[buf][ar * LDT + ak] = pa;
#pragma unroll
    for (int j = 0; j < 8; ++j) {
      v4bf p;
#pragma unroll
      for (int r = 0; r < 4; ++r) p[r] = (__bf16)wreg[r][j];
      *(v4bf*)&Bs[buf][(bn + j) * LDT + kb] = p;  // transposed: Bs[n][k]
    }
  };

  load_stage(0);
  store_stage(0);
  __syncthreads();

  const int rsel = lane & 15;
  const int koff = (lane >> 4) * 8;
  int cur = 0;

  for (int k0 = 0; k0 < D_; k0 += TK) {
    const bool more = (k0 + TK) < D_;
    if (more) load_stage(k0 + TK);   // HBM latency hidden under WMMA below
#pragma unroll
    for (int sb = 0; sb < 2; ++sb) {
      const int ks = sb * 32;
      v16bf a   = load_frag(&As[cur][(wm * 16 + rsel) * LDT + ks], koff);
      v16bf b0  = load_frag(&Bs[cur][(wn * 32 + rsel) * LDT + ks], koff);
      v16bf b1f = load_frag(&Bs[cur][(wn * 32 + 16 + rsel) * LDT + ks], koff);
      acc0 = __builtin_amdgcn_wmma_f32_16x16x32_bf16(false, a, false, b0,
                                                     (short)0, acc0, false, false);
      acc1 = __builtin_amdgcn_wmma_f32_16x16x32_bf16(false, a, false, b1f,
                                                     (short)0, acc1, false, false);
    }
    if (more) {
      store_stage(cur ^ 1);   // co-executes with the WMMAs above
      __syncthreads();        // single barrier per K-step
    }
    cur ^= 1;
  }

  const int ncol = nBase + wn * 32 + (lane & 15);
  const int mtop = (lane >> 4) * 8;
  const float bia0 = b1[e * H_ + ncol];
  const float bia1 = b1[e * H_ + ncol + 16];
#pragma unroll
  for (int r = 0; r < 8; ++r) {
    int m = wm * 16 + mtop + r;
    if (rowBase + m < cnt) {
      __bf16* hp = h + (size_t)s_row[m] * H_;
      hp[ncol]      = (__bf16)fmaxf(acc0[r] + bia0, 0.0f);
      hp[ncol + 16] = (__bf16)fmaxf(acc1[r] + bia1, 0.0f);
    }
  }
}

// ---------------------------------------------------------------------------
// Stage B: out[tok] += gate * (h_row @ W2[e] + b2[e])
// A-tiles (already bf16) DMA'd via global_load_async_to_lds_b128 into the
// double-buffered LDS tile; B-tiles register-staged + converted.  One
// barrier per K-step; s_wait_asynccnt 0 before the barrier publishes the DMA.
__global__ __launch_bounds__(256)
void expert_down_kernel(const unsigned short* __restrict__ h_raw,
                        const float* __restrict__ W2,
                        const float* __restrict__ b2,
                        const int* __restrict__ counts,
                        const int* __restrict__ tokslot,
                        const float* __restrict__ gatearr,
                        float* __restrict__ out) {
  const int e = blockIdx.z;
  const int cnt = counts[e];
  const int rowBase = blockIdx.y * TM;
  if (rowBase >= cnt) return;
  const int nBase = blockIdx.x * TN;   // over D
  const int tid = threadIdx.x;
  const int lane = tid & 31, wave = tid >> 5;
  const int wm = wave >> 2, wn = wave & 3;

  const __bf16* __restrict__ h = (const __bf16*)h_raw;
  __shared__ __attribute__((aligned(16))) __bf16 As[2][TM * LDT];
  __shared__ __attribute__((aligned(16))) __bf16 Bs[2][TN * LDT];
  __shared__ int s_row[TM];
  __shared__ float s_gate[TM];

  if (tid < TM) {
    int r = imin(rowBase + tid, cnt - 1);
    s_row[tid] = tokslot[e * B_ + r];
    s_gate[tid] = gatearr[e * B_ + r];
  }
  __syncthreads();

  v8f acc0 = {}, acc1 = {};
  const float* __restrict__ W = W2 + (size_t)e * H_ * D_;

  const int ar = tid >> 3;
  const int ak = (tid & 7) * 8;
  const __bf16* __restrict__ hrow = h + (size_t)s_row[ar] * H_;
  const int kb = (tid >> 4) * 4;
  const int bn = (tid & 15) * 8;

  float wreg[4][8];

  auto stage_a_async = [&](int buf, int k0) {
    uint32_t lds_off = (uint32_t)(uintptr_t)&As[buf][ar * LDT + ak];
    async_copy_b128(lds_off, (uint64_t)(uintptr_t)(hrow + k0 + ak));
  };
  auto load_b = [&](int k0) {
#pragma unroll
    for (int r = 0; r < 4; ++r) {
      const float* src = W + (size_t)(k0 + kb + r) * D_ + nBase + bn;
#pragma unroll
      for (int j = 0; j < 8; ++j) wreg[r][j] = src[j];
    }
  };
  auto store_b = [&](int buf) {
#pragma unroll
    for (int j = 0; j < 8; ++j) {
      v4bf p;
#pragma unroll
      for (int r = 0; r < 4; ++r) p[r] = (__bf16)wreg[r][j];
      *(v4bf*)&Bs[buf][(bn + j) * LDT + kb] = p;
    }
  };

  stage_a_async(0, 0);
  load_b(0);
  store_b(0);
  wait_async();
  __syncthreads();

  const int rsel = lane & 15;
  const int koff = (lane >> 4) * 8;
  int cur = 0;

  for (int k0 = 0; k0 < H_; k0 += TK) {
    const bool more = (k0 + TK) < H_;
    if (more) {
      stage_a_async(cur ^ 1, k0 + TK);  // DMA next A tile while we compute
      load_b(k0 + TK);
    }
#pragma unroll
    for (int sb = 0; sb < 2; ++sb) {
      const int ks = sb * 32;
      v16bf a   = load_frag(&As[cur][(wm * 16 + rsel) * LDT + ks], koff);
      v16bf b0  = load_frag(&Bs[cur][(wn * 32 + rsel) * LDT + ks], koff);
      v16bf b1f = load_frag(&Bs[cur][(wn * 32 + 16 + rsel) * LDT + ks], koff);
      acc0 = __builtin_amdgcn_wmma_f32_16x16x32_bf16(false, a, false, b0,
                                                     (short)0, acc0, false, false);
      acc1 = __builtin_amdgcn_wmma_f32_16x16x32_bf16(false, a, false, b1f,
                                                     (short)0, acc1, false, false);
    }
    if (more) {
      store_b(cur ^ 1);   // co-executes with WMMA
      wait_async();       // our DMA into As[cur^1] complete before barrier
      __syncthreads();
    }
    cur ^= 1;
  }

  const int ncol = nBase + wn * 32 + (lane & 15);
  const int mtop = (lane >> 4) * 8;
  const float bia0 = b2[e * D_ + ncol];
  const float bia1 = b2[e * D_ + ncol + 16];
#pragma unroll
  for (int r = 0; r < 8; ++r) {
    int m = wm * 16 + mtop + r;
    if (rowBase + m < cnt) {
      int tok = s_row[m] >> 1;
      float g = s_gate[m];
      atomicAdd(out + (size_t)tok * D_ + ncol,      g * (acc0[r] + bia0));
      atomicAdd(out + (size_t)tok * D_ + ncol + 16, g * (acc1[r] + bia1));
    }
  }
}

// ---------------------------------------------------------------------------
extern "C" void kernel_launch(void* const* d_in, const int* in_sizes, int n_in,
                              void* d_out, int out_size, void* d_ws, size_t ws_size,
                              hipStream_t stream) {
  (void)in_sizes; (void)n_in; (void)out_size; (void)ws_size;
  const float* x     = (const float*)d_in[0];
  const float* gamma = (const float*)d_in[1];
  const float* beta  = (const float*)d_in[2];
  const float* keys  = (const float*)d_in[3];
  const float* W1    = (const float*)d_in[4];
  const float* b1    = (const float*)d_in[5];
  const float* W2    = (const float*)d_in[6];
  const float* b2    = (const float*)d_in[7];
  float* out = (float*)d_out;

  // Workspace layout: counts | tokslot | gates | h (bf16, [B*2][H]) ~= 34 MB
  char* ws = (char*)d_ws;
  int*   counts  = (int*)ws;
  int*   tokslot = (int*)(ws + 256);
  float* gatearr = (float*)(ws + 256 + (size_t)E_ * B_ * 4);
  unsigned short* h = (unsigned short*)(ws + 256 + 2 * (size_t)E_ * B_ * 4);

  zero_kernel<<<2048, 256, 0, stream>>>(out, B_ * D_, counts);
  gate_kernel<<<B_, 256, 0, stream>>>(x, gamma, beta, keys, counts, tokslot, gatearr);

  dim3 gA(H_ / TN, B_ / TM, E_);   // 32 x 64 x 16 (most row-tiles exit early)
  expert_up_kernel<<<gA, 256, 0, stream>>>(x, W1, b1, counts, tokslot, h);

  dim3 gB(D_ / TN, B_ / TM, E_);   // 8 x 64 x 16
  expert_down_kernel<<<gB, 256, 0, stream>>>(h, W2, b2, counts, tokslot, gatearr, out);
}
